// ChargeAwarePotential_23622320128114
// MI455X (gfx1250) — matrix operand order repaired and checked by reading, not compiled, not run on hardware
//
#include <hip/hip_runtime.h>
#include <hip/hip_bf16.h>

#define NATOM  4096
#define HID    64
#define NELEM  119
#define K_E    14.3996f

typedef __attribute__((ext_vector_type(16))) _Float16 v16h;
typedef __attribute__((ext_vector_type(8)))  float    v8f;

union AFrag { v16h v; _Float16 e[16]; uint4 u[2]; };

#define WMMA_F16(Afrag, Bfrag, Cacc) \
    __builtin_amdgcn_wmma_f32_16x16x32_f16(false, (Afrag), false, (Bfrag), (short)0, (Cacc), false, false)

__device__ __forceinline__ float softplusf(float x) {
    return (x > 20.f) ? x : log1pf(__expf(x));
}

// ---------------------------------------------------------------------------
// Pass 0: zero the scalar accumulators (E lives in d_out[0]).
__global__ void init_kernel(float* E_out, float* qsum, float* csum) {
    if (threadIdx.x == 0) { *E_out = 0.f; *qsum = 0.f; *csum = 0.f; }
}

// ---------------------------------------------------------------------------
// Pass 1: h0 = embed[Z] + q_embed[Q+5] + s_embed[S-1]; f32, f16, f16-transposed.
__global__ void h0_kernel(const int* __restrict__ Z, const int* __restrict__ Q,
                          const int* __restrict__ S,
                          const float* __restrict__ embed,
                          const float* __restrict__ q_embed,
                          const float* __restrict__ s_embed,
                          float* __restrict__ h0f, _Float16* __restrict__ h0h,
                          _Float16* __restrict__ h0T) {
    int idx = blockIdx.x * blockDim.x + threadIdx.x;
    if (idx >= NATOM * HID) return;
    int i = idx >> 6, cc = idx & 63;
    float v = embed[Z[i] * HID + cc] + q_embed[(Q[0] + 5) * HID + cc]
            + s_embed[(S[0] - 1) * HID + cc];
    h0f[idx] = v;
    h0h[idx] = (_Float16)v;
    h0T[(size_t)cc * NATOM + i] = (_Float16)v;
}

// ---------------------------------------------------------------------------
// Pass 1b: f16 copies of the 64x64 head weights, row-major and transposed.
__global__ void wprep_kernel(const float* __restrict__ eW1, const float* __restrict__ cW1,
                             _Float16* __restrict__ eW1h, _Float16* __restrict__ cW1h,
                             _Float16* __restrict__ eW1T, _Float16* __restrict__ cW1T) {
    int idx = blockIdx.x * blockDim.x + threadIdx.x;
    if (idx >= HID * HID) return;
    int k = idx >> 6, cc = idx & 63;
    float e = eW1[idx], cw = cW1[idx];
    eW1h[idx] = (_Float16)e;
    cW1h[idx] = (_Float16)cw;
    eW1T[(size_t)cc * HID + k] = (_Float16)e;
    cW1T[(size_t)cc * HID + k] = (_Float16)cw;
}

// ---------------------------------------------------------------------------
// Pass 2: h = h0 + P @ h0 (f16 out), P_ij = exp(-r_ij/2) built in-register.
__global__ __launch_bounds__(32)
void mp_gemm_kernel(const float* __restrict__ R, const float* __restrict__ h0f,
                    const _Float16* __restrict__ h0T, _Float16* __restrict__ hh) {
    const int lane = threadIdx.x;
    const int g  = lane >> 4;
    const int mn = lane & 15;
    const int i0 = blockIdx.x << 4;
    const int i  = i0 + mn;
    const float rix = R[i*3+0], riy = R[i*3+1], riz = R[i*3+2];

    __shared__ float Rs[32][3];

    v8f acc0 = {}, acc1 = {}, acc2 = {}, acc3 = {};

    for (int k0 = 0; k0 < NATOM; k0 += 32) {
        Rs[lane][0] = R[(k0+lane)*3+0];
        Rs[lane][1] = R[(k0+lane)*3+1];
        Rs[lane][2] = R[(k0+lane)*3+2];
        __syncthreads();

        AFrag A;
#pragma unroll
        for (int t = 0; t < 8; ++t) {
            int k1 = 16*g + t, k2 = 16*g + 8 + t;
            float dx = rix - Rs[k1][0], dy = riy - Rs[k1][1], dz = riz - Rs[k1][2];
            float r1 = sqrtf(dx*dx + dy*dy + dz*dz);
            A.e[t] = (k0 + k1 == i) ? (_Float16)0.f : (_Float16)__expf(-0.5f * r1);
            dx = rix - Rs[k2][0]; dy = riy - Rs[k2][1]; dz = riz - Rs[k2][2];
            float r2 = sqrtf(dx*dx + dy*dy + dz*dz);
            A.e[8+t] = (k0 + k2 == i) ? (_Float16)0.f : (_Float16)__expf(-0.5f * r2);
        }

        v16h b0 = *(const v16h*)(h0T + (size_t)( 0 + mn) * NATOM + k0 + 16*g);
        v16h b1 = *(const v16h*)(h0T + (size_t)(16 + mn) * NATOM + k0 + 16*g);
        v16h b2 = *(const v16h*)(h0T + (size_t)(32 + mn) * NATOM + k0 + 16*g);
        v16h b3 = *(const v16h*)(h0T + (size_t)(48 + mn) * NATOM + k0 + 16*g);

        acc0 = WMMA_F16(A.v, b0, acc0);
        acc1 = WMMA_F16(A.v, b1, acc1);
        acc2 = WMMA_F16(A.v, b2, acc2);
        acc3 = WMMA_F16(A.v, b3, acc3);
        __syncthreads();
    }

#pragma unroll
    for (int v = 0; v < 8; ++v) {
        int row = i0 + v + 8*g;
        hh[(size_t)row*HID +  0 + mn] = (_Float16)(h0f[row*HID +  0 + mn] + acc0[v]);
        hh[(size_t)row*HID + 16 + mn] = (_Float16)(h0f[row*HID + 16 + mn] + acc1[v]);
        hh[(size_t)row*HID + 32 + mn] = (_Float16)(h0f[row*HID + 32 + mn] + acc2[v]);
        hh[(size_t)row*HID + 48 + mn] = (_Float16)(h0f[row*HID + 48 + mn] + acc3[v]);
    }
}

// ---------------------------------------------------------------------------
// Pass 3: both heads as WMMA GEMMs. One wave per 16-atom tile:
// U = h_tile @ eW1 + b1, V = h_tile @ cW1 + b1 (16x64 each, f32 accum), then
// the silu/silu' epilogue directly on the C-fragment layout (row=v+8g, col=16t+n).
__global__ __launch_bounds__(256)
void heads_wmma_kernel(const _Float16* __restrict__ hh,
                       const _Float16* __restrict__ eW1T, const _Float16* __restrict__ cW1T,
                       const float* __restrict__ eb1, const float* __restrict__ cb1,
                       const float* __restrict__ eW2, const float* __restrict__ eb2,
                       const float* __restrict__ cW2, const float* __restrict__ cb2,
                       _Float16* __restrict__ g1h, _Float16* __restrict__ g2h,
                       float* __restrict__ q_raw, float* __restrict__ E_out,
                       float* __restrict__ qsum) {
    const int wave = threadIdx.x >> 5, lane = threadIdx.x & 31;
    const int g = lane >> 4, n = lane & 15;
    const int i0 = (blockIdx.x * 8 + wave) << 4;

    const _Float16* arow = hh + (size_t)(i0 + n) * HID;
    AFrag A0, A1;
    A0.u[0] = *(const uint4*)(arow +  0 + 16*g); A0.u[1] = *(const uint4*)(arow +  8 + 16*g);
    A1.u[0] = *(const uint4*)(arow + 32 + 16*g); A1.u[1] = *(const uint4*)(arow + 40 + 16*g);

    v8f zerov = {};
    v8f accE[4], accC[4];
#pragma unroll
    for (int t = 0; t < 4; ++t) { accE[t] = zerov; accC[t] = zerov; }

#pragma unroll
    for (int t = 0; t < 4; ++t) {
        const _Float16* be = eW1T + (size_t)(16*t + n) * HID;
        const _Float16* bc = cW1T + (size_t)(16*t + n) * HID;
        v16h be0 = *(const v16h*)(be +  0 + 16*g);
        v16h be1 = *(const v16h*)(be + 32 + 16*g);
        v16h bc0 = *(const v16h*)(bc +  0 + 16*g);
        v16h bc1 = *(const v16h*)(bc + 32 + 16*g);
        accE[t] = WMMA_F16(A0.v, be0, accE[t]);
        accE[t] = WMMA_F16(A1.v, be1, accE[t]);
        accC[t] = WMMA_F16(A0.v, bc0, accC[t]);
        accC[t] = WMMA_F16(A1.v, bc1, accC[t]);
    }

    float Ep = 0.f;
    float qrow[8];
#pragma unroll
    for (int v = 0; v < 8; ++v) qrow[v] = 0.f;

#pragma unroll
    for (int t = 0; t < 4; ++t) {
        const int col = 16*t + n;
        const float be = eb1[col], bc = cb1[col];
        const float we = eW2[col], wc = cW2[col];
#pragma unroll
        for (int v = 0; v < 8; ++v) {
            const int row = i0 + v + 8*g;
            float u  = accE[t][v] + be;
            float su = 1.f/(1.f + __expf(-u));
            Ep += u*su*we;
            g1h[(size_t)row*HID + col] = (_Float16)(su*(1.f + u*(1.f - su))*we);
            float vv = accC[t][v] + bc;
            float sv = 1.f/(1.f + __expf(-vv));
            qrow[v] += vv*sv*wc;
            g2h[(size_t)row*HID + col] = (_Float16)(sv*(1.f + vv*(1.f - sv))*wc);
        }
    }

    for (int off = 16; off; off >>= 1) Ep += __shfl_xor(Ep, off, 32);
    if (lane == 0) atomicAdd(E_out, Ep + 16.f*eb2[0]);

    float qloc = 0.f;
#pragma unroll
    for (int v = 0; v < 8; ++v) {
        float s = qrow[v];
        for (int off = 8; off; off >>= 1) s += __shfl_xor(s, off, 32);
        if (n == 0) {
            float qr = s + cb2[0];
            q_raw[i0 + v + 8*g] = qr;
            qloc += qr;
        }
    }
    if (n == 0) atomicAdd(qsum, qloc);
}

// ---------------------------------------------------------------------------
// Pass 4: charge conservation.
__global__ void qfix_kernel(const float* __restrict__ q_raw, const float* __restrict__ qsum,
                            const int* __restrict__ Q, float* __restrict__ q) {
    int i = blockIdx.x * blockDim.x + threadIdx.x;
    if (i < NATOM) q[i] = q_raw[i] + ((float)Q[0] - qsum[0]) * (1.f / (float)NATOM);
}

// ---------------------------------------------------------------------------
// Pass 5: Coulomb energy + c_i = K_E * sum_{j!=i} q_j / s_ij (one wave per row).
__global__ __launch_bounds__(256)
void coulomb_kernel(const float* __restrict__ R, const int* __restrict__ Z,
                    const float* __restrict__ q, const float* __restrict__ shield,
                    float* __restrict__ c, float* __restrict__ csum,
                    float* __restrict__ E_out) {
    const int wave = threadIdx.x >> 5, lane = threadIdx.x & 31;
    const int i = blockIdx.x * 8 + wave;
    const float rix = R[i*3+0], riy = R[i*3+1], riz = R[i*3+2];
    const float qi = q[i];
    const int zi = Z[i] * NELEM;
    float ci = 0.f, ec = 0.f;
    for (int j = lane; j < NATOM; j += 32) {
        if (j == i) continue;
        float dx = rix - R[j*3+0], dy = riy - R[j*3+1], dz = riz - R[j*3+2];
        float r2 = dx*dx + dy*dy + dz*dz;
        float a  = softplusf(shield[zi + Z[j]]);
        float s  = sqrtf(r2 + a*a);
        float t  = q[j] / s;
        ci += t;
        if (j > i) ec += qi * t;
    }
    for (int off = 16; off; off >>= 1) {
        ci += __shfl_xor(ci, off, 32);
        ec += __shfl_xor(ec, off, 32);
    }
    if (lane == 0) {
        float cv = K_E * ci;
        c[i] = cv;
        atomicAdd(csum, cv);
        atomicAdd(E_out, K_E * ec);
    }
}

// ---------------------------------------------------------------------------
// Pass 6: D = g1 @ eW1^T + diag(chat) * (g2 @ cW1^T) as two WMMA GEMMs.
// B[c][k] = eW1[k*64+c] -> row k of eW1h is contiguous in c: direct v16h loads.
__global__ __launch_bounds__(256)
void dgrad_wmma_kernel(const _Float16* __restrict__ g1h, const _Float16* __restrict__ g2h,
                       const _Float16* __restrict__ eW1h, const _Float16* __restrict__ cW1h,
                       const float* __restrict__ c, const float* __restrict__ csum,
                       _Float16* __restrict__ Dh) {
    const int wave = threadIdx.x >> 5, lane = threadIdx.x & 31;
    const int g = lane >> 4, n = lane & 15;
    const int i0 = (blockIdx.x * 8 + wave) << 4;

    const _Float16* a1 = g1h + (size_t)(i0 + n) * HID;
    const _Float16* a2 = g2h + (size_t)(i0 + n) * HID;
    AFrag A10, A11, A20, A21;
    A10.u[0] = *(const uint4*)(a1 +  0 + 16*g); A10.u[1] = *(const uint4*)(a1 +  8 + 16*g);
    A11.u[0] = *(const uint4*)(a1 + 32 + 16*g); A11.u[1] = *(const uint4*)(a1 + 40 + 16*g);
    A20.u[0] = *(const uint4*)(a2 +  0 + 16*g); A20.u[1] = *(const uint4*)(a2 +  8 + 16*g);
    A21.u[0] = *(const uint4*)(a2 + 32 + 16*g); A21.u[1] = *(const uint4*)(a2 + 40 + 16*g);

    v8f zerov = {};
    v8f acc1[4], acc2[4];
#pragma unroll
    for (int t = 0; t < 4; ++t) { acc1[t] = zerov; acc2[t] = zerov; }

#pragma unroll
    for (int t = 0; t < 4; ++t) {
        const _Float16* b1 = eW1h + (size_t)(16*t + n) * HID;
        const _Float16* b2 = cW1h + (size_t)(16*t + n) * HID;
        v16h b10 = *(const v16h*)(b1 +  0 + 16*g);
        v16h b11 = *(const v16h*)(b1 + 32 + 16*g);
        v16h b20 = *(const v16h*)(b2 +  0 + 16*g);
        v16h b21 = *(const v16h*)(b2 + 32 + 16*g);
        acc1[t] = WMMA_F16(A10.v, b10, acc1[t]);
        acc1[t] = WMMA_F16(A11.v, b11, acc1[t]);
        acc2[t] = WMMA_F16(A20.v, b20, acc2[t]);
        acc2[t] = WMMA_F16(A21.v, b21, acc2[t]);
    }

    const float cbar = csum[0] * (1.f / (float)NATOM);
    float chat[8];
#pragma unroll
    for (int v = 0; v < 8; ++v) chat[v] = c[i0 + v + 8*g] - cbar;

#pragma unroll
    for (int t = 0; t < 4; ++t) {
#pragma unroll
        for (int v = 0; v < 8; ++v) {
            const int row = i0 + v + 8*g;
            Dh[(size_t)row*HID + 16*t + n] = (_Float16)(acc1[t][v] + chat[v]*acc2[t][v]);
        }
    }
}

// ---------------------------------------------------------------------------
// Pass 7: forces. Per 16x16 pair tile: T = D_i . h0_j, S = h0_i . D_j via WMMA,
// then w_ij = -P/2*(T+S) - K_E qi qj r / s^3, F_i = -sum w*(Ri-Rj)/r.
__global__ __launch_bounds__(256)
void force_kernel(const float* __restrict__ R, const int* __restrict__ Z,
                  const float* __restrict__ q, const float* __restrict__ shield,
                  const _Float16* __restrict__ h0h, const _Float16* __restrict__ Dh,
                  float* __restrict__ out) {
    const int tid  = threadIdx.x;
    const int wave = tid >> 5, lane = tid & 31;
    const int g = lane >> 4, n = lane & 15;
    const int i0 = blockIdx.x << 4;

    __shared__ float Ris[16][3];
    __shared__ float qis[16];
    __shared__ int   Zis[16];
    __shared__ float facc[16][3];

    if (tid < 48) facc[tid/3][tid%3] = 0.f;
    if (tid < 16) {
        Ris[tid][0] = R[(i0+tid)*3+0];
        Ris[tid][1] = R[(i0+tid)*3+1];
        Ris[tid][2] = R[(i0+tid)*3+2];
        qis[tid] = q[i0+tid];
        Zis[tid] = Z[i0+tid] * NELEM;
    }
    __syncthreads();

    const _Float16* aT = Dh  + (size_t)(i0 + n) * HID;
    const _Float16* aS = h0h + (size_t)(i0 + n) * HID;
    AFrag AT0, AT1, AS0, AS1;
    AT0.u[0] = *(const uint4*)(aT +  0 + 16*g); AT0.u[1] = *(const uint4*)(aT +  8 + 16*g);
    AT1.u[0] = *(const uint4*)(aT + 32 + 16*g); AT1.u[1] = *(const uint4*)(aT + 40 + 16*g);
    AS0.u[0] = *(const uint4*)(aS +  0 + 16*g); AS0.u[1] = *(const uint4*)(aS +  8 + 16*g);
    AS1.u[0] = *(const uint4*)(aS + 32 + 16*g); AS1.u[1] = *(const uint4*)(aS + 40 + 16*g);

    float fx[8], fy[8], fz[8];
#pragma unroll
    for (int v = 0; v < 8; ++v) { fx[v] = 0.f; fy[v] = 0.f; fz[v] = 0.f; }

    for (int jt = wave; jt < NATOM/16; jt += 8) {
        const int j = jt*16 + n;
        const float rjx = R[j*3+0], rjy = R[j*3+1], rjz = R[j*3+2];
        const float qj = q[j];
        const int   zj = Z[j];
        const _Float16* bT = h0h + (size_t)j * HID;
        const _Float16* bS = Dh  + (size_t)j * HID;
        {
            int jn = (jt + 8)*16 + n;
            if (jn < NATOM) {
                __builtin_prefetch(h0h + (size_t)jn * HID, 0, 1);
                __builtin_prefetch(Dh  + (size_t)jn * HID, 0, 1);
            }
        }
        v16h bT0 = *(const v16h*)(bT +  0 + 16*g);
        v16h bT1 = *(const v16h*)(bT + 32 + 16*g);
        v16h bS0 = *(const v16h*)(bS +  0 + 16*g);
        v16h bS1 = *(const v16h*)(bS + 32 + 16*g);

        v8f T = {};
        T = WMMA_F16(AT0.v, bT0, T);
        T = WMMA_F16(AT1.v, bT1, T);
        v8f Sm = {};
        Sm = WMMA_F16(AS0.v, bS0, Sm);
        Sm = WMMA_F16(AS1.v, bS1, Sm);

#pragma unroll
        for (int v = 0; v < 8; ++v) {
            const int mi = v + 8*g;
            const int i  = i0 + mi;
            if (i == j) continue;
            float dx = Ris[mi][0] - rjx, dy = Ris[mi][1] - rjy, dz = Ris[mi][2] - rjz;
            float r2 = dx*dx + dy*dy + dz*dz;
            float r  = sqrtf(r2);
            float p  = __expf(-0.5f * r);
            float a  = softplusf(shield[Zis[mi] + zj]);
            float s2 = r2 + a*a;
            float s  = sqrtf(s2);
            float w  = -0.5f * p * (T[v] + Sm[v]) - K_E * qis[mi] * qj * r / (s2 * s);
            float inv = w / r;
            fx[v] += inv * dx; fy[v] += inv * dy; fz[v] += inv * dz;
        }
    }
#pragma unroll
    for (int v = 0; v < 8; ++v) {
        const int mi = v + 8*g;
        atomicAdd(&facc[mi][0], fx[v]);
        atomicAdd(&facc[mi][1], fy[v]);
        atomicAdd(&facc[mi][2], fz[v]);
    }
    __syncthreads();
    if (tid < 48) out[1 + (i0 + tid/3)*3 + tid%3] = -facc[tid/3][tid%3];
}

// ---------------------------------------------------------------------------
extern "C" void kernel_launch(void* const* d_in, const int* in_sizes, int n_in,
                              void* d_out, int out_size, void* d_ws, size_t ws_size,
                              hipStream_t stream) {
    (void)in_sizes; (void)n_in; (void)out_size; (void)ws_size;
    const int*   Z      = (const int*)  d_in[0];
    const float* R      = (const float*)d_in[1];
    const int*   Q      = (const int*)  d_in[2];
    const int*   S      = (const int*)  d_in[3];
    const float* embed  = (const float*)d_in[4];
    const float* q_emb  = (const float*)d_in[5];
    const float* s_emb  = (const float*)d_in[6];
    const float* eW1    = (const float*)d_in[7];
    const float* eb1    = (const float*)d_in[8];
    const float* eW2    = (const float*)d_in[9];
    const float* eb2    = (const float*)d_in[10];
    const float* cW1    = (const float*)d_in[11];
    const float* cb1    = (const float*)d_in[12];
    const float* cW2    = (const float*)d_in[13];
    const float* cb2    = (const float*)d_in[14];
    const float* shield = (const float*)d_in[15];
    float* out = (float*)d_out;

    char* p = (char*)d_ws;
    auto take = [&](size_t bytes) -> char* {
        char* r = p; p += (bytes + 255) & ~(size_t)255; return r;
    };
    float*    h0f   = (float*)   take((size_t)NATOM*HID*sizeof(float));
    _Float16* h0h   = (_Float16*)take((size_t)NATOM*HID*sizeof(_Float16));
    _Float16* h0T   = (_Float16*)take((size_t)HID*NATOM*sizeof(_Float16));
    _Float16* hh    = (_Float16*)take((size_t)NATOM*HID*sizeof(_Float16));
    _Float16* g1h   = (_Float16*)take((size_t)NATOM*HID*sizeof(_Float16));
    _Float16* g2h   = (_Float16*)take((size_t)NATOM*HID*sizeof(_Float16));
    _Float16* Dh    = (_Float16*)take((size_t)NATOM*HID*sizeof(_Float16));
    _Float16* eW1h  = (_Float16*)take((size_t)HID*HID*sizeof(_Float16));
    _Float16* cW1h  = (_Float16*)take((size_t)HID*HID*sizeof(_Float16));
    _Float16* eW1T  = (_Float16*)take((size_t)HID*HID*sizeof(_Float16));
    _Float16* cW1T  = (_Float16*)take((size_t)HID*HID*sizeof(_Float16));
    float*    q_raw = (float*)   take((size_t)NATOM*sizeof(float));
    float*    qv    = (float*)   take((size_t)NATOM*sizeof(float));
    float*    cv    = (float*)   take((size_t)NATOM*sizeof(float));
    float*    scal  = (float*)   take(256);
    float* qsum = scal + 0;
    float* csum = scal + 1;

    init_kernel<<<1, 32, 0, stream>>>(out, qsum, csum);
    h0_kernel<<<(NATOM*HID + 255)/256, 256, 0, stream>>>(Z, Q, S, embed, q_emb, s_emb,
                                                         h0f, h0h, h0T);
    wprep_kernel<<<(HID*HID + 255)/256, 256, 0, stream>>>(eW1, cW1, eW1h, cW1h, eW1T, cW1T);
    mp_gemm_kernel<<<NATOM/16, 32, 0, stream>>>(R, h0f, h0T, hh);
    heads_wmma_kernel<<<NATOM/128, 256, 0, stream>>>(hh, eW1T, cW1T, eb1, cb1,
                                                     eW2, eb2, cW2, cb2,
                                                     g1h, g2h, q_raw, out, qsum);
    qfix_kernel<<<(NATOM + 255)/256, 256, 0, stream>>>(q_raw, qsum, Q, qv);
    coulomb_kernel<<<NATOM/8, 256, 0, stream>>>(R, Z, qv, shield, cv, csum, out);
    dgrad_wmma_kernel<<<NATOM/128, 256, 0, stream>>>(g1h, g2h, eW1h, cW1h, cv, csum, Dh);
    force_kernel<<<NATOM/16, 256, 0, stream>>>(R, Z, qv, shield, h0h, Dh, out);
}
